// ShiftReduceCompressor_48009144434909
// MI455X (gfx1250) — compile-verified, matrix operand order checked by reading, not run on hardware
//
#include <hip/hip_runtime.h>
#include <hip/hip_bf16.h>
#include <math.h>

// ---------------- model constants ----------------
#define BATCH   64
#define SEQL    64
#define DMODEL  128
#define NHEAD   4
#define HD      32
#define NLAYER  2
#define DOUT    64
#define MAXW    32
#define NITER   32
#define DFF     512

// LDS strides (halves), all row strides are multiples of 8 halves (16 B)
#define XH_STR   136   // x f16 rows
#define QKV_STR  416   // qkv f16 rows (q:0..127 k:128..255 v:256..383)
#define FH_STR   528   // ffn hidden rows

typedef __attribute__((ext_vector_type(16))) _Float16 v16h;
typedef __attribute__((ext_vector_type(8)))  float    v8f;

union Frag {
  v16h     v;
  _Float16 h[16];
  uint4    q[2];
};

struct Params {
  const int*           token_ids;
  const unsigned char* padding_mask;
  const float *tok_emb, *pos_emb;
  const float *bqkv, *bo, *ln1g, *ln1b, *b1f, *b2f, *ln2g, *ln2b;
  const float *proj_W, *proj_b;
  const float *gru_Wih, *gru_Whh, *gru_bih, *gru_bhh;
  const float *acc_W1, *acc_b1, *acc_W2, *acc_b2;
  const float *emt_W1, *emt_b1, *emt_W2, *emt_b2;
  const float *evt_W1, *evt_b1, *evt_W2, *evt_b2;
  const _Float16 *Wqkv_h, *Wo_h, *W1_h, *W2_h;
  float* out_t1; float* out_lp; float* out_dec; int* out_emit;
};

struct SM {
  alignas(16) float     x[MAXW][DMODEL];     // f32 activations (residual stream)
  alignas(16) _Float16  xh[MAXW][XH_STR];    // f16 copy / attn-output staging
  alignas(16) float     gru_h[DMODEL];
  int   win_ids[MAXW];
  int   win_len, input_pos, done_flag, nt_tok, seq_len, emit_cnt;
  float norm_red;
  float logits[3];
  union {
    struct { alignas(16) _Float16 qkv[MAXW][QKV_STR];
             alignas(16) float    s[2][MAXW][MAXW]; } att;      // ~34 KB
    struct { alignas(16) _Float16 fh[MAXW][FH_STR]; } ffn;      // ~33 KB
    struct { alignas(16) float vec[DOUT]; float t1[DOUT];
             float gi[3*DMODEL]; float gh[3*DMODEL];
             float gnew[DMODEL]; float hidden[DMODEL];
             float nemb[DMODEL]; float cat[2*DMODEL + DOUT]; } post;
  } u;
};

__device__ __forceinline__ float sigm(float x) { return 1.0f / (1.0f + __expf(-x)); }
__device__ __forceinline__ float logsig(float x) {
  return (x < 0.0f) ? (x - log1pf(__expf(x))) : (-log1pf(__expf(-x)));
}
__device__ __forceinline__ float rng01(unsigned long long x) {
  x ^= x >> 33; x *= 0xff51afd7ed558ccdULL;
  x ^= x >> 33; x *= 0xc4ceb9fe1a85ec53ULL;
  x ^= x >> 33;
  return (float)(x >> 40) * (1.0f / 16777216.0f);
}

// C[32 x N] = act(A[32 x K] @ W^T + bias)  (A in LDS f16, W global f16 [N][K])
// mode 0: Cf += v (f32 residual accumulate); mode 1: f16 store; mode 2: f16 relu store
// `wave` must be an SGPR value (readfirstlane'd) so the tile loop is scalar and
// EXEC stays all-ones around every WMMA, as the ISA requires.
__device__ __forceinline__ void gemm_tiles(
    const _Float16* __restrict__ A, int lda,
    const _Float16* __restrict__ W, int K, int N,
    const float* __restrict__ bias, int mode,
    float* Cf, int ldcf, _Float16* Ch, int ldch,
    int wave, int lane) {
  const int hi = lane >> 4, ln = lane & 15;
  const int T = 2 * (N >> 4);
  for (int t = wave; t < T; t += 8) {
    const int Mt = t & 1, Nt = t >> 1;
    const int col = Nt * 16 + ln;
    const float bv = bias[col];
    v8f acc;
    #pragma unroll
    for (int r = 0; r < 8; ++r) acc[r] = bv;
    const _Float16* arow = A + (Mt * 16 + ln) * lda;
    const _Float16* brow = W + (size_t)col * K;
    __builtin_prefetch(brow, 0, 3);   // WGP-scope prefetch of this wave's B row
    for (int kt = 0; kt < K; kt += 32) {
      Frag a, b;
      a.q[0] = *(const uint4*)(arow + kt + 8 * hi);
      a.q[1] = *(const uint4*)(arow + kt + 16 + 8 * hi);
      b.q[0] = *(const uint4*)(brow + kt + 8 * hi);
      b.q[1] = *(const uint4*)(brow + kt + 16 + 8 * hi);
      acc = __builtin_amdgcn_wmma_f32_16x16x32_f16(false, a.v, false, b.v,
                                                   (short)0, acc, false, false);
    }
    const int row0 = Mt * 16 + hi * 8;
    #pragma unroll
    for (int r = 0; r < 8; ++r) {
      float v = acc[r];
      if (mode == 2) v = fmaxf(v, 0.0f);
      if (mode == 0) Cf[(row0 + r) * ldcf + col] += v;
      else           Ch[(row0 + r) * ldch + col] = (_Float16)v;
    }
  }
}

__device__ void layernorm_rows(SM& sm, const float* g, const float* b, int tid) {
  if (tid < MAXW) {
    float* row = sm.x[tid];
    float m = 0.0f;
    for (int k = 0; k < DMODEL; ++k) m += row[k];
    m *= (1.0f / DMODEL);
    float v = 0.0f;
    for (int k = 0; k < DMODEL; ++k) { float d = row[k] - m; v += d * d; }
    v *= (1.0f / DMODEL);
    const float inv = rsqrtf(v + 1e-5f);
    for (int k = 0; k < DMODEL; ++k) {
      float y = (row[k] - m) * inv * g[k] + b[k];
      row[k] = y;
      sm.xh[tid][k] = (_Float16)y;
    }
  }
}

__global__ void cvt_f32_to_f16_kernel(const float* __restrict__ src,
                                      _Float16* __restrict__ dst, int n) {
  int i = blockIdx.x * blockDim.x + threadIdx.x;
  if (i < n) dst[i] = (_Float16)src[i];
}

__global__ void __launch_bounds__(256, 1)
ShiftReduceCompressor_main(Params P) {
  __shared__ SM sm;
  const int n    = blockIdx.x;          // batch element
  const int tid  = threadIdx.x;
  // Wave index as a compiler-visible scalar: keeps the GEMM tile loops in
  // scalar branches (EXEC all-ones at every WMMA) instead of EXEC-masked
  // vector loops.
  const int wave = __builtin_amdgcn_readfirstlane(tid >> 5);
  const int lane = tid & 31;
  const int hi   = lane >> 4, ln = lane & 15;

  // ---------------- init ----------------
  if (tid < MAXW)  sm.win_ids[tid] = 0;
  if (tid < DMODEL) sm.gru_h[tid]  = 0.0f;
  if (tid == 0) {
    int sl = 0;
    for (int j = 0; j < SEQL; ++j) sl += (P.padding_mask[n * SEQL + j] == 0);
    sm.seq_len = sl; sm.win_len = 0; sm.input_pos = 0;
    sm.done_flag = (sl == 0); sm.emit_cnt = 0;
  }
  __syncthreads();

  for (int it = 0; it < NITER; ++it) {
    // ------------- embeddings -------------
    for (int i = tid; i < MAXW * DMODEL; i += 256) {
      const int w = i >> 7, d = i & 127;
      const float v = P.tok_emb[(size_t)sm.win_ids[w] * DMODEL + d]
                    + P.pos_emb[w * DMODEL + d];
      sm.x[w][d]  = v;
      sm.xh[w][d] = (_Float16)v;
    }
    __syncthreads();
    const int  winlen0 = sm.win_len;
    const bool active  = (sm.done_flag == 0);

    // ------------- encoder (2 layers, all GEMMs via WMMA) -------------
    for (int l = 0; l < NLAYER; ++l) {
      gemm_tiles(&sm.xh[0][0], XH_STR, P.Wqkv_h + (size_t)l * 3 * DMODEL * DMODEL,
                 DMODEL, 3 * DMODEL, P.bqkv + l * 3 * DMODEL, 1,
                 nullptr, 0, &sm.u.att.qkv[0][0], QKV_STR, wave, lane);
      __syncthreads();

      for (int hg = 0; hg < 2; ++hg) {
        const int hl   = wave >> 2;            // 0..1: head within group (SGPR)
        const int head = hg * 2 + hl;
        const int sub  = wave & 3, Mt = sub & 1, Nt = sub >> 1;
        // scores tile: S = Q K^T / sqrt(32)
        {
          Frag a, b; v8f acc;
          #pragma unroll
          for (int r = 0; r < 8; ++r) acc[r] = 0.0f;
          const _Float16* arow = &sm.u.att.qkv[Mt * 16 + ln][head * HD];
          const _Float16* brow = &sm.u.att.qkv[Nt * 16 + ln][DMODEL + head * HD];
          a.q[0] = *(const uint4*)(arow + 8 * hi);
          a.q[1] = *(const uint4*)(arow + 16 + 8 * hi);
          b.q[0] = *(const uint4*)(brow + 8 * hi);
          b.q[1] = *(const uint4*)(brow + 16 + 8 * hi);
          acc = __builtin_amdgcn_wmma_f32_16x16x32_f16(false, a.v, false, b.v,
                                                       (short)0, acc, false, false);
          const int row0 = Mt * 16 + hi * 8;
          #pragma unroll
          for (int r = 0; r < 8; ++r)
            sm.u.att.s[hl][row0 + r][Nt * 16 + ln] = acc[r] * 0.176776695f;
        }
        __syncthreads();
        // masked softmax over keys (rows handled by threads 0..63)
        if (tid < 64) {
          const int h2 = tid >> 5, row = tid & 31;
          float* sr = sm.u.att.s[h2][row];
          float mx = -1e30f;
          for (int j = 0; j < MAXW; ++j) {
            const bool valid = (j == 0) || ((j < winlen0) && active);
            const float v = sr[j] + (valid ? 0.0f : -1e9f);
            sr[j] = v;
            mx = fmaxf(mx, v);
          }
          float ssum = 0.0f;
          for (int j = 0; j < MAXW; ++j) { float e = __expf(sr[j] - mx); sr[j] = e; ssum += e; }
          const float inv = 1.0f / ssum;
          for (int j = 0; j < MAXW; ++j) sr[j] *= inv;
        }
        __syncthreads();
        // O = P V  -> staged into xh (f16)
        {
          Frag a, b; v8f acc;
          #pragma unroll
          for (int r = 0; r < 8; ++r) acc[r] = 0.0f;
          const float* prow = sm.u.att.s[hl][Mt * 16 + ln];
          const float4 p0 = *(const float4*)(prow + 8 * hi);
          const float4 p1 = *(const float4*)(prow + 8 * hi + 4);
          const float4 p2 = *(const float4*)(prow + 8 * hi + 16);
          const float4 p3 = *(const float4*)(prow + 8 * hi + 20);
          a.h[0] = (_Float16)p0.x; a.h[1] = (_Float16)p0.y; a.h[2] = (_Float16)p0.z; a.h[3] = (_Float16)p0.w;
          a.h[4] = (_Float16)p1.x; a.h[5] = (_Float16)p1.y; a.h[6] = (_Float16)p1.z; a.h[7] = (_Float16)p1.w;
          a.h[8] = (_Float16)p2.x; a.h[9] = (_Float16)p2.y; a.h[10]= (_Float16)p2.z; a.h[11]= (_Float16)p2.w;
          a.h[12]= (_Float16)p3.x; a.h[13]= (_Float16)p3.y; a.h[14]= (_Float16)p3.z; a.h[15]= (_Float16)p3.w;
          const int dcol = head * HD + Nt * 16 + ln;
          #pragma unroll
          for (int e = 0; e < 16; ++e) {
            const int key = (e < 8) ? (e + 8 * hi) : (e + 8 + 8 * hi);
            b.h[e] = sm.u.att.qkv[key][2 * DMODEL + dcol];
          }
          acc = __builtin_amdgcn_wmma_f32_16x16x32_f16(false, a.v, false, b.v,
                                                       (short)0, acc, false, false);
          const int row0 = Mt * 16 + hi * 8;
          #pragma unroll
          for (int r = 0; r < 8; ++r)
            sm.xh[row0 + r][dcol] = (_Float16)acc[r];
        }
        __syncthreads();
      }

      // Wo: residual-accumulate into f32 x
      gemm_tiles(&sm.xh[0][0], XH_STR, P.Wo_h + (size_t)l * DMODEL * DMODEL,
                 DMODEL, DMODEL, P.bo + l * DMODEL, 0,
                 &sm.x[0][0], DMODEL, nullptr, 0, wave, lane);
      __syncthreads();
      layernorm_rows(sm, P.ln1g + l * DMODEL, P.ln1b + l * DMODEL, tid);
      __syncthreads();
      // FFN
      gemm_tiles(&sm.xh[0][0], XH_STR, P.W1_h + (size_t)l * DFF * DMODEL,
                 DMODEL, DFF, P.b1f + l * DFF, 2,
                 nullptr, 0, &sm.u.ffn.fh[0][0], FH_STR, wave, lane);
      __syncthreads();
      gemm_tiles(&sm.u.ffn.fh[0][0], FH_STR, P.W2_h + (size_t)l * DMODEL * DFF,
                 DFF, DMODEL, P.b2f + l * DMODEL, 0,
                 &sm.x[0][0], DMODEL, nullptr, 0, wave, lane);
      __syncthreads();
      layernorm_rows(sm, P.ln2g + l * DMODEL, P.ln2b + l * DMODEL, tid);
      __syncthreads();
    }

    // ------------- projection + normalize -------------
    const int lastrow = max(winlen0 - 1, 0);
    if (tid < DOUT) {
      float acc = P.proj_b[tid];
      const float* w  = P.proj_W + tid * DMODEL;
      const float* xr = sm.x[lastrow];
      for (int k = 0; k < DMODEL; ++k) acc += w[k] * xr[k];
      sm.u.post.vec[tid] = acc;
    }
    __syncthreads();
    if (tid == 0) {
      float ss = 0.0f;
      for (int o = 0; o < DOUT; ++o) ss += sm.u.post.vec[o] * sm.u.post.vec[o];
      sm.norm_red = fmaxf(sqrtf(ss), 1e-12f);
    }
    __syncthreads();
    const bool nonempty = (winlen0 > 0) && active;
    if (tid < DOUT) {
      const float t = nonempty ? sm.u.post.vec[tid] / sm.norm_red : 0.0f;
      sm.u.post.t1[tid] = t;
      P.out_t1[((size_t)it * BATCH + n) * DOUT + tid] = t;
    }
    __syncthreads();

    // ------------- GRU -------------
    for (int k = tid; k < 3 * DMODEL; k += 256) {
      float a1 = P.gru_bih[k];
      const float* w1 = P.gru_Wih + (size_t)k * DOUT;
      for (int j = 0; j < DOUT; ++j) a1 += w1[j] * sm.u.post.t1[j];
      sm.u.post.gi[k] = a1;
      float a2 = P.gru_bhh[k];
      const float* w2 = P.gru_Whh + (size_t)k * DMODEL;
      for (int j = 0; j < DMODEL; ++j) a2 += w2[j] * sm.gru_h[j];
      sm.u.post.gh[k] = a2;
    }
    __syncthreads();
    if (tid < DMODEL) {
      const float r = sigm(sm.u.post.gi[tid] + sm.u.post.gh[tid]);
      const float z = sigm(sm.u.post.gi[DMODEL + tid] + sm.u.post.gh[DMODEL + tid]);
      const float nn = tanhf(sm.u.post.gi[2 * DMODEL + tid] + r * sm.u.post.gh[2 * DMODEL + tid]);
      const float hnew = (1.0f - z) * nn + z * sm.gru_h[tid];
      sm.u.post.gnew[tid] = active ? hnew : sm.gru_h[tid];
    }
    __syncthreads();
    if (tid < DMODEL) sm.gru_h[tid] = sm.u.post.gnew[tid];

    // ------------- next-token embedding -------------
    if (tid == 0) {
      const int cl = min(max(sm.input_pos, 0), SEQL - 1);
      sm.nt_tok = P.token_ids[n * SEQL + cl];
    }
    __syncthreads();
    const bool has_input = active && (sm.input_pos < sm.seq_len);
    if (tid < DMODEL)
      sm.u.post.nemb[tid] = has_input ? P.tok_emb[(size_t)sm.nt_tok * DMODEL + tid] : 0.0f;
    __syncthreads();

    // ------------- 3 MLP heads -------------
    for (int hidx = 0; hidx < 3; ++hidx) {
      const int insz = (hidx == 0) ? (2 * DMODEL + DOUT) : (DMODEL + DOUT);
      const float* W1h = (hidx == 0) ? P.acc_W1 : (hidx == 1) ? P.emt_W1 : P.evt_W1;
      const float* b1h = (hidx == 0) ? P.acc_b1 : (hidx == 1) ? P.emt_b1 : P.evt_b1;
      const float* W2h = (hidx == 0) ? P.acc_W2 : (hidx == 1) ? P.emt_W2 : P.evt_W2;
      const float* b2h = (hidx == 0) ? P.acc_b2 : (hidx == 1) ? P.emt_b2 : P.evt_b2;
      for (int i = tid; i < insz; i += 256) {
        float v;
        if (hidx == 0)
          v = (i < DMODEL) ? sm.u.post.nemb[i]
            : (i < DMODEL + DOUT) ? sm.u.post.t1[i - DMODEL]
            : sm.gru_h[i - DMODEL - DOUT];
        else
          v = (i < DOUT) ? sm.u.post.t1[i] : sm.gru_h[i - DOUT];
        sm.u.post.cat[i] = v;
      }
      __syncthreads();
      if (tid < DMODEL) {
        float a = b1h[tid];
        const float* w = W1h + (size_t)tid * insz;
        for (int j = 0; j < insz; ++j) a += w[j] * sm.u.post.cat[j];
        sm.u.post.hidden[tid] = fmaxf(a, 0.0f);
      }
      __syncthreads();
      if (tid == 0) {
        float l2 = b2h[0];
        for (int hh = 0; hh < DMODEL; ++hh) l2 += W2h[hh] * sm.u.post.hidden[hh];
        sm.logits[hidx] = l2;
      }
      __syncthreads();
    }

    // ------------- sampling + bookkeeping (thread 0) -------------
    if (tid == 0) {
      const float al = sm.logits[0], el = sm.logits[1], vl = sm.logits[2];
      const unsigned long long base =
          0x2545F4914F6CDD1DULL + 42ULL * 0x9E3779B97F4A7C15ULL
          + (unsigned long long)it * 0x94D049BB133111EBULL
          + (unsigned long long)n  * 0xBF58476D1CE4E5B9ULL;
      const int as = rng01(base + 1) < sigm(al);
      const int es = rng01(base + 2) < sigm(el);
      const int vs = rng01(base + 3) < sigm(vl);
      const size_t ob = ((size_t)it * BATCH + n) * 3;
      P.out_lp[ob + 0] = logsig(as ? al : -al);
      P.out_lp[ob + 1] = logsig(es ? el : -el);
      P.out_lp[ob + 2] = logsig(vs ? vl : -vl);
      P.out_dec[ob + 0] = (float)as;
      P.out_dec[ob + 1] = (float)es;
      P.out_dec[ob + 2] = (float)vs;

      bool a = as && has_input;
      bool e = es && nonempty;
      bool v = vs && nonempty;
      const bool none = active && !(a || e || v);
      a = a || (none && has_input);
      e = e || (none && !has_input && (sm.win_len > 0));
      const bool a_eff = a && (sm.win_len < MAXW);
      if (a_eff) {
        sm.win_ids[sm.win_len] = sm.nt_tok;
        sm.input_pos += 1;
        sm.win_len   += 1;
      }
      const bool emit  = e && (sm.win_len > 0);
      const bool v_eff = v && (sm.win_len > 0);
      if (v_eff) {
        const int first = sm.win_ids[0];
        for (int j = 0; j < MAXW - 1; ++j) sm.win_ids[j] = sm.win_ids[j + 1];
        sm.win_ids[MAXW - 1] = first;
        sm.win_len -= 1;
      }
      if (active && (sm.input_pos >= sm.seq_len) && (sm.win_len == 0))
        sm.done_flag = 1;
      sm.emit_cnt += (int)emit;
    }
    __syncthreads();
  }

  if (tid == 0) P.out_emit[n] = sm.emit_cnt;
}

extern "C" void kernel_launch(void* const* d_in, const int* in_sizes, int n_in,
                              void* d_out, int out_size, void* d_ws, size_t ws_size,
                              hipStream_t stream) {
  (void)in_sizes; (void)n_in; (void)out_size; (void)ws_size;

  // f16 weight staging in workspace
  _Float16* ws = (_Float16*)d_ws;
  const int nQKV = NLAYER * 3 * DMODEL * DMODEL;  // 98304
  const int nWO  = NLAYER * DMODEL * DMODEL;      // 32768
  const int nW1  = NLAYER * DFF * DMODEL;         // 131072
  const int nW2  = NLAYER * DMODEL * DFF;         // 131072
  _Float16* wqkv = ws;
  _Float16* wo   = wqkv + nQKV;
  _Float16* w1   = wo   + nWO;
  _Float16* w2   = w1   + nW1;

  cvt_f32_to_f16_kernel<<<(nQKV + 255) / 256, 256, 0, stream>>>((const float*)d_in[6],  wqkv, nQKV);
  cvt_f32_to_f16_kernel<<<(nWO  + 255) / 256, 256, 0, stream>>>((const float*)d_in[8],  wo,   nWO);
  cvt_f32_to_f16_kernel<<<(nW1  + 255) / 256, 256, 0, stream>>>((const float*)d_in[12], w1,   nW1);
  cvt_f32_to_f16_kernel<<<(nW2  + 255) / 256, 256, 0, stream>>>((const float*)d_in[14], w2,   nW2);

  Params P;
  P.token_ids    = (const int*)d_in[0];
  P.padding_mask = (const unsigned char*)d_in[1];
  P.tok_emb = (const float*)d_in[4];   P.pos_emb = (const float*)d_in[5];
  P.bqkv = (const float*)d_in[7];      P.bo   = (const float*)d_in[9];
  P.ln1g = (const float*)d_in[10];     P.ln1b = (const float*)d_in[11];
  P.b1f  = (const float*)d_in[13];     P.b2f  = (const float*)d_in[15];
  P.ln2g = (const float*)d_in[16];     P.ln2b = (const float*)d_in[17];
  P.proj_W = (const float*)d_in[18];   P.proj_b = (const float*)d_in[19];
  P.gru_Wih = (const float*)d_in[20];  P.gru_Whh = (const float*)d_in[21];
  P.gru_bih = (const float*)d_in[22];  P.gru_bhh = (const float*)d_in[23];
  P.acc_W1 = (const float*)d_in[24];   P.acc_b1 = (const float*)d_in[25];
  P.acc_W2 = (const float*)d_in[26];   P.acc_b2 = (const float*)d_in[27];
  P.emt_W1 = (const float*)d_in[28];   P.emt_b1 = (const float*)d_in[29];
  P.emt_W2 = (const float*)d_in[30];   P.emt_b2 = (const float*)d_in[31];
  P.evt_W1 = (const float*)d_in[32];   P.evt_b1 = (const float*)d_in[33];
  P.evt_W2 = (const float*)d_in[34];   P.evt_b2 = (const float*)d_in[35];
  P.Wqkv_h = wqkv; P.Wo_h = wo; P.W1_h = w1; P.W2_h = w2;

  float* out = (float*)d_out;
  P.out_t1  = out;                                         // [32,64,64]
  P.out_lp  = out + (size_t)NITER * BATCH * DOUT;          // [32,64,3]
  P.out_dec = P.out_lp + (size_t)NITER * BATCH * 3;        // [32,64,3]
  P.out_emit = (int*)(P.out_dec + (size_t)NITER * BATCH * 3); // [64] int32

  ShiftReduceCompressor_main<<<BATCH, 256, 0, stream>>>(P);
}
